// SelfAttention_v1_57518202028173
// MI455X (gfx1250) — matrix operand order repaired and checked by reading, not compiled, not run on hardware
//
#include <hip/hip_runtime.h>
#include <hip/hip_bf16.h>

// MI455X / gfx1250, wave32. bf16 WMMA with f32 accumulate.
// Key design points (reasoned from MI455X specs, compile-only loop):
//  - 48MB of bf16 Q/K/V fits the 192MB L2 -> stream matrix fragments
//    directly from global (L2-resident); LDS only where a layout change
//    is unavoidable.
//  - V is stored pre-transposed by the projection kernel so the attention
//    inner loop needs no cooperative staging and no block barriers.
//  - global_prefetch_b8 (via __builtin_prefetch) hides K/V^T stream latency.

typedef __attribute__((ext_vector_type(16))) __bf16 v16bf;
typedef __attribute__((ext_vector_type(8)))  __bf16 v8bf;
typedef __attribute__((ext_vector_type(8)))  float  v8f;

#define N_ROWS 8192
#define D_DIM  1024

__device__ __forceinline__ v8f zero8() {
  v8f z = {0.f, 0.f, 0.f, 0.f, 0.f, 0.f, 0.f, 0.f};
  return z;
}

__device__ __forceinline__ v16bf join8(v8bf lo, v8bf hi) {
  return __builtin_shufflevector(lo, hi, 0, 1, 2, 3, 4, 5, 6, 7,
                                         8, 9, 10, 11, 12, 13, 14, 15);
}

__device__ __forceinline__ v8f wmma_bf16(v16bf a, v16bf b, v8f c) {
  // 8 args: (neg_a, A, neg_b, B, c_mod, C, reuse_a, reuse_b)
  return __builtin_amdgcn_wmma_f32_16x16x32_bf16(false, a, false, b,
                                                 (short)0, c, false, false);
}

// ---------------------------------------------------------------------------
// Kernel 1: Y = X @ W for W in {w_q, w_k, w_v}  (blockIdx.z selects)
// X: [8192,1024] f32, W: [1024,1024] f32 -> Q,K: [8192,1024] bf16 row-major,
// V: stored TRANSPOSED as V^T [1024,8192] bf16 (so the attention kernel can
// load V B-fragments as contiguous 32B reads with no LDS staging).
// Block tile 128x128, 256 threads (8 waves), each wave = 16 rows x 128 cols.
// ---------------------------------------------------------------------------
__global__ void __launch_bounds__(256)
qkv_gemm_kernel(const float* __restrict__ x,
                const float* __restrict__ wq,
                const float* __restrict__ wk,
                const float* __restrict__ wv,
                __bf16* __restrict__ qb,
                __bf16* __restrict__ kb,
                __bf16* __restrict__ vbt) {
  // Staging LDS is dead after the k-loop's final barrier; reuse it for the
  // V-transpose bounce so total LDS stays at 32KB.
  __shared__ union SMem {
    struct { __bf16 Xs[128][40]; __bf16 Wst[128][48]; } stage;
    __bf16 Ct[128][128];
  } smem;

  const float* w;
  __bf16* out;
  if (blockIdx.z == 0)      { w = wq; out = qb; }
  else if (blockIdx.z == 1) { w = wk; out = kb; }
  else                      { w = wv; out = vbt; }

  const int tid  = threadIdx.x;
  const int wave = tid >> 5;
  const int lane = tid & 31;
  const int lh   = lane & 15;  // index within 16-lane half
  const int half = lane >> 4;  // which half of the wave

  const int mbase = blockIdx.x * 128;
  const int nbase = blockIdx.y * 128;

  v8f acc[8];
#pragma unroll
  for (int t = 0; t < 8; ++t) acc[t] = zero8();

  for (int k0 = 0; k0 < D_DIM; k0 += 32) {
    // Stage X tile (128x32) -> bf16 LDS. Coalesced along k.
#pragma unroll
    for (int i = 0; i < 16; ++i) {
      int e = tid + 256 * i;
      int m = e >> 5, k = e & 31;
      smem.stage.Xs[m][k] = (__bf16)x[(size_t)(mbase + m) * D_DIM + k0 + k];
    }
    // Stage W tile (32x128) transposed -> bf16 LDS. Coalesced along n.
#pragma unroll
    for (int i = 0; i < 16; ++i) {
      int e = tid + 256 * i;
      int k = e >> 7, n = e & 127;
      smem.stage.Wst[n][k] = (__bf16)w[(size_t)(k0 + k) * D_DIM + nbase + n];
    }
    __syncthreads();

    // A fragment (16x32 bf16): lane holds row lh; lanes 0-15 K=[0..15],
    // lanes 16-31 K=[16..31] (contiguous, matching B).
    const __bf16* arow = &smem.stage.Xs[wave * 16 + lh][0];
    v8bf alo = *(const v8bf*)(arow + half * 16);
    v8bf ahi = *(const v8bf*)(arow + half * 16 + 8);
    v16bf a = join8(alo, ahi);

#pragma unroll
    for (int nt = 0; nt < 8; ++nt) {
      // B fragment (32x16): lane holds column nt*16+lh, 16 contiguous K
      // starting at half*16 (contiguous k order, matching A). 32B aligned (stride 48).
      v16bf b = *(const v16bf*)&smem.stage.Wst[nt * 16 + lh][half * 16];
      acc[nt] = wmma_bf16(a, b, acc[nt]);
    }
    __syncthreads();
  }

  // C layout: VGPR v, lanes 0-15 -> M=v, lanes 16-31 -> M=v+8; N=lh.
  if (blockIdx.z != 2) {
    // Q, K: row-major bf16 store.
#pragma unroll
    for (int nt = 0; nt < 8; ++nt) {
#pragma unroll
      for (int v = 0; v < 8; ++v) {
        int row = mbase + wave * 16 + v + 8 * half;
        int col = nbase + nt * 16 + lh;
        out[(size_t)row * D_DIM + col] = (__bf16)acc[nt][v];
      }
    }
  } else {
    // V: transpose through LDS, then coalesced v8bf stores to V^T [D][N].
#pragma unroll
    for (int nt = 0; nt < 8; ++nt)
#pragma unroll
      for (int v = 0; v < 8; ++v)
        smem.Ct[nt * 16 + lh][wave * 16 + v + 8 * half] = (__bf16)acc[nt][v];
    __syncthreads();
#pragma unroll
    for (int i = 0; i < 8; ++i) {
      int e = tid + 256 * i;   // 2048 chunks of 8 bf16
      int c = e >> 4;          // 0..127 : output column (= V^T row)
      int r = (e & 15) * 8;    // 0..120 : key offset within tile
      v8bf val = *(const v8bf*)&smem.Ct[c][r];
      *(v8bf*)(out + (size_t)(nbase + c) * N_ROWS + mbase + r) = val;
    }
  }
}

// ---------------------------------------------------------------------------
// Kernel 2: flash attention. O = softmax(Q K^T / 32) V
// Q,K row-major; V^T [1024][8192]; all bf16, L2-resident. Out f32.
// grid.x: 64 bands of 128 Q rows; grid.y: 8 chunks of 128 output cols
// (scores recomputed per chunk so the O accumulator fits in registers:
//  16 rows x 128 cols f32 per wave = 64 VGPRs/lane).
// No block barriers in the main loop: all WMMA operands come straight from
// L2-resident global memory except the per-wave P re-layout bounce.
// ---------------------------------------------------------------------------
__global__ void __launch_bounds__(256)
flash_attn_kernel(const __bf16* __restrict__ qb,
                  const __bf16* __restrict__ kb,
                  const __bf16* __restrict__ vbt,
                  float* __restrict__ out) {
  __shared__ __bf16 Plds[8][16][32];   // per-wave P relayout buffer (8KB)

  const int tid  = threadIdx.x;
  const int wave = tid >> 5;
  const int lane = tid & 31;
  const int lh   = lane & 15;
  const int half = lane >> 4;

  const int qrow0 = blockIdx.x * 128 + wave * 16;
  const int c0    = blockIdx.y * 128;  // output / V column chunk

  const float scale = 0.03125f;  // 1/sqrt(1024)

  v8f o[8];
#pragma unroll
  for (int t = 0; t < 8; ++t) o[t] = zero8();
  float m_run[8], l_run[8];
#pragma unroll
  for (int v = 0; v < 8; ++v) { m_run[v] = -3.0e38f; l_run[v] = 0.f; }

  const __bf16* qrowp = qb + (size_t)(qrow0 + lh) * D_DIM;

  for (int j = 0; j < N_ROWS; j += 32) {
    // Prefetch next key tile of K and next V^T segment (global_prefetch_b8).
    if (j + 32 < N_ROWS) {
      __builtin_prefetch(kb + (size_t)(j + 32 + lh) * D_DIM + half * 16, 0, 3);
      __builtin_prefetch(kb + (size_t)(j + 48 + lh) * D_DIM + half * 16, 0, 3);
      __builtin_prefetch(vbt + (size_t)(c0 + lane) * N_ROWS + j + 32, 0, 3);
      __builtin_prefetch(vbt + (size_t)(c0 + 64 + lane) * N_ROWS + j + 32, 0, 3);
    }

    // ---- S = (Q K^T) for 16 rows x 32 keys, contraction over D=1024 ----
    v8f s0 = zero8(), s1 = zero8();
    for (int d0 = 0; d0 < D_DIM; d0 += 32) {
      v8bf alo = *(const v8bf*)(qrowp + d0 + half * 16);
      v8bf ahi = *(const v8bf*)(qrowp + d0 + half * 16 + 8);
      v16bf a = join8(alo, ahi);
      // B = K^T: column n is K row (j+n), contiguous d -> direct 32B loads.
      v16bf b0 = *(const v16bf*)(kb + (size_t)(j + lh) * D_DIM + d0 + half * 16);
      v16bf b1 = *(const v16bf*)(kb + (size_t)(j + 16 + lh) * D_DIM + d0 + half * 16);
      s0 = wmma_bf16(a, b0, s0);
      s1 = wmma_bf16(a, b1, s1);
    }

    // ---- online softmax update (rows live in 16-lane halves) ----
    float alpha[8];
#pragma unroll
    for (int v = 0; v < 8; ++v) {
      float sv0 = s0[v] * scale;
      float sv1 = s1[v] * scale;
      float mt = fmaxf(sv0, sv1);
#pragma unroll
      for (int mk = 1; mk <= 8; mk <<= 1)
        mt = fmaxf(mt, __shfl_xor(mt, mk, 32));
      float mn = fmaxf(m_run[v], mt);
      float a_ = __expf(m_run[v] - mn);
      float p0 = __expf(sv0 - mn);
      float p1 = __expf(sv1 - mn);
      float rs = p0 + p1;
#pragma unroll
      for (int mk = 1; mk <= 8; mk <<= 1)
        rs += __shfl_xor(rs, mk, 32);
      l_run[v] = l_run[v] * a_ + rs;
      m_run[v] = mn;
      alpha[v] = a_;
      s0[v] = p0;
      s1[v] = p1;
    }
#pragma unroll
    for (int t = 0; t < 8; ++t)
#pragma unroll
      for (int v = 0; v < 8; ++v) o[t][v] *= alpha[v];

    // ---- P: C-layout -> A-layout via per-wave LDS bounce (wave-local) ----
#pragma unroll
    for (int v = 0; v < 8; ++v) {
      Plds[wave][v + 8 * half][lh]      = (__bf16)s0[v];
      Plds[wave][v + 8 * half][16 + lh] = (__bf16)s1[v];
    }
    const __bf16* prow = &Plds[wave][lh][0];
    v8bf plo = *(const v8bf*)(prow + half * 16);
    v8bf phi = *(const v8bf*)(prow + half * 16 + 8);
    v16bf ap = join8(plo, phi);

    // ---- O += P @ V_tile ; B-fragments straight from L2-resident V^T ----
#pragma unroll
    for (int nt = 0; nt < 8; ++nt) {
      v16bf bv = *(const v16bf*)(vbt + (size_t)(c0 + nt * 16 + lh) * N_ROWS +
                                 j + half * 16);
      o[nt] = wmma_bf16(ap, bv, o[nt]);
    }
  }

  // ---- normalize and store f32 output ----
#pragma unroll
  for (int v = 0; v < 8; ++v) l_run[v] = 1.f / l_run[v];
#pragma unroll
  for (int nt = 0; nt < 8; ++nt)
#pragma unroll
    for (int v = 0; v < 8; ++v) {
      int row = qrow0 + v + 8 * half;
      int col = c0 + nt * 16 + lh;
      out[(size_t)row * D_DIM + col] = o[nt][v] * l_run[v];
    }
}

// ---------------------------------------------------------------------------
extern "C" void kernel_launch(void* const* d_in, const int* in_sizes, int n_in,
                              void* d_out, int out_size, void* d_ws, size_t ws_size,
                              hipStream_t stream) {
  (void)in_sizes; (void)n_in; (void)out_size; (void)ws_size;

  const float* x  = (const float*)d_in[0];
  const float* wq = (const float*)d_in[1];
  const float* wk = (const float*)d_in[2];
  const float* wv = (const float*)d_in[3];
  float* out = (float*)d_out;

  const size_t mat = (size_t)N_ROWS * D_DIM;  // 8M elements
  __bf16* qb  = (__bf16*)d_ws;                // 16 MB, Q row-major
  __bf16* kb  = qb + mat;                     // 16 MB, K row-major
  __bf16* vbt = kb + mat;                     // 16 MB, V^T [1024][8192]

  dim3 g1(N_ROWS / 128, D_DIM / 128, 3);
  qkv_gemm_kernel<<<g1, 256, 0, stream>>>(x, wq, wk, wv, qb, kb, vbt);

  dim3 g2(N_ROWS / 128, D_DIM / 128);
  flash_attn_kernel<<<g2, 256, 0, stream>>>(qb, kb, vbt, out);
}
